// Attention_89335319756981
// MI455X (gfx1250) — matrix operand order repaired and checked by reading, compiled, not run on hardware
//
#include <hip/hip_runtime.h>
#include <hip/hip_bf16.h>

#define DIM   768
#define HEADS 12
#define HD    64
#define NB    8
#define HW    576
#define TFR   4
#define NSEQ  2880            // HW + TFR*HW
#define MROWS (NB * NSEQ)     // 23040
#define PROWS (NB * HW)       // 4608

typedef __attribute__((ext_vector_type(16))) __bf16 v16bf;
typedef __attribute__((ext_vector_type(8)))  float  v8f;
typedef __attribute__((ext_vector_type(4)))  __bf16 v4bf;

// ---- fragment helpers (layouts per CDNA5 ISA 7.12.2) ----
__device__ __forceinline__ v16bf frag16(const __bf16* p0, const __bf16* p1) {
  union { uint4 u; __bf16 h[8]; } a, b;
  a.u = *reinterpret_cast<const uint4*>(p0);
  b.u = *reinterpret_cast<const uint4*>(p1);
  v16bf r;
#pragma unroll
  for (int i = 0; i < 8; ++i) { r[i] = a.h[i]; r[i + 8] = b.h[i]; }
  return r;
}
__device__ __forceinline__ v8f bwmma(v16bf a, v16bf b, v8f c) {
  return __builtin_amdgcn_wmma_f32_16x16x32_bf16(false, a, false, b, (short)0, c,
                                                 false, false);
}
__device__ __forceinline__ float wave_max(float v) {
#pragma unroll
  for (int m = 16; m >= 1; m >>= 1) v = fmaxf(v, __shfl_xor(v, m, 32));
  return v;
}
__device__ __forceinline__ float wave_sum(float v) {
#pragma unroll
  for (int m = 16; m >= 1; m >>= 1) v += __shfl_xor(v, m, 32);
  return v;
}

// ---- CDNA5 async global->LDS copy (ASYNCcnt path, ISA ch.10/15.18) ----
__device__ __forceinline__ void async_ld16(const void* g, void* l) {
  unsigned loff = (unsigned)(unsigned long long)l;      // LDS addr = flat[31:0]
  unsigned long long ga = (unsigned long long)g;
  asm volatile("global_load_async_to_lds_b128 %0, %1, off"
               :: "v"(loff), "v"(ga) : "memory");
}
__device__ __forceinline__ void wait_async0() {
  asm volatile("s_wait_asynccnt 0" ::: "memory");
}

// ---- fp32 -> bf16 (vectorized by 4) ----
__global__ void cvt_bf16(const float* __restrict__ src, __bf16* __restrict__ dst, int n4) {
  int i = blockIdx.x * blockDim.x + threadIdx.x;
  if (i >= n4) return;
  float4 f = reinterpret_cast<const float4*>(src)[i];
  v4bf o;
  o[0] = (__bf16)f.x; o[1] = (__bf16)f.y; o[2] = (__bf16)f.z; o[3] = (__bf16)f.w;
  reinterpret_cast<v4bf*>(dst)[i] = o;
}

// ---- passthrough: out[b, HW:, :] = x[b, HW:, :]  (same flat offsets) ----
__global__ void copy_tail(const float* __restrict__ x, float* __restrict__ out) {
  const int per_b = TFR * HW * DIM / 4;  // float4 per batch
  int i = blockIdx.x * blockDim.x + threadIdx.x;
  if (i >= NB * per_b) return;
  int b = i / per_b, r = i % per_b;
  size_t off = ((size_t)b * NSEQ + HW) * (DIM / 4) + r;
  reinterpret_cast<float4*>(out)[off] = reinterpret_cast<const float4*>(x)[off];
}

// ================= GEMM core: 64x128 tile, 8 waves, 2x2 blocking ============
// A: [Mrows x 768] bf16 row-major. W: [Ncols x 768] bf16 row-major (acts as B^T).
// Double-buffered LDS filled with global_load_async_to_lds_b128.
struct GemmFrag {
  v8f acc[2][2];
};

template <typename StoreFn>
__device__ __forceinline__ void gemm_64x128(const __bf16* __restrict__ A,
                                            const __bf16* __restrict__ W,
                                            int row0, int col0, StoreFn&& store) {
  __shared__ alignas(16) __bf16 Abuf[2][64 * 32];
  __shared__ alignas(16) __bf16 Bbuf[2][128 * 32];
  const int wv = threadIdx.x >> 5, lane = threadIdx.x & 31;
  const int wm = wv >> 2, wn = wv & 3;            // 2 x 4 waves, each 32x32
  const int m = lane & 15, half = lane >> 4;

  auto stage = [&](int bufi, int kt) {
    const int k0 = kt * 32;
    {  // A tile: 64 rows x 32 k = 256 x 16B chunks, 1 per thread
      int c = threadIdx.x, r = c >> 2, o = (c & 3) * 8;
      async_ld16(A + (size_t)(row0 + r) * DIM + k0 + o, &Abuf[bufi][r * 32 + o]);
    }
#pragma unroll
    for (int j = 0; j < 2; ++j) {  // B tile: 128 rows x 32 k = 512 chunks, 2/thread
      int c = threadIdx.x + j * 256, r = c >> 2, o = (c & 3) * 8;
      async_ld16(W + (size_t)(col0 + r) * DIM + k0 + o, &Bbuf[bufi][r * 32 + o]);
    }
  };

  v8f acc[2][2] = {};
  stage(0, 0);
  wait_async0();
  __syncthreads();
  for (int kt = 0; kt < 24; ++kt) {
    const int cur = kt & 1;
    if (kt + 1 < 24) stage(1 - cur, kt + 1);
    v16bf af[2], bf[2];
#pragma unroll
    for (int s = 0; s < 2; ++s) {
      const __bf16* ar = &Abuf[cur][(wm * 32 + s * 16 + m) * 32];
      af[s] = frag16(ar + half * 8, ar + 16 + half * 8);
      const __bf16* br = &Bbuf[cur][(wn * 32 + s * 16 + m) * 32];
      bf[s] = frag16(br + half * 8, br + 16 + half * 8);
    }
#pragma unroll
    for (int sm = 0; sm < 2; ++sm)
#pragma unroll
      for (int sn = 0; sn < 2; ++sn)
        acc[sm][sn] = bwmma(af[sm], bf[sn], acc[sm][sn]);
    wait_async0();
    __syncthreads();
  }
  // store: output element (row0 + wm*32 + sm*16 + r + 8*half, col0 + wn*32 + sn*16 + m)
#pragma unroll
  for (int sm = 0; sm < 2; ++sm)
#pragma unroll
    for (int sn = 0; sn < 2; ++sn) {
      const int cg = col0 + wn * 32 + sn * 16 + m;
#pragma unroll
      for (int r = 0; r < 8; ++r) {
        int rg = row0 + wm * 32 + sm * 16 + r + 8 * half;
        store(rg, cg, acc[sm][sn][r]);
      }
    }
}

// ---- QKV GEMM: [23040x768] x [768x2304]; scatter to q (scaled) / k / vT ----
__global__ void gemm_qkv(const __bf16* __restrict__ X, const __bf16* __restrict__ W,
                         __bf16* __restrict__ qb, __bf16* __restrict__ kb,
                         __bf16* __restrict__ vT) {
  const int row0 = blockIdx.x * 64, col0 = blockIdx.y * 128;
  gemm_64x128(X, W, row0, col0, [&](int rg, int cg, float v) {
    const int sec = cg / DIM, rem = cg % DIM;
    const int h = rem >> 6, d = rem & 63;
    const int b_ = rg / NSEQ, n = rg % NSEQ;
    const size_t bh = (size_t)b_ * HEADS + h;
    if (sec == 0) {
      if (n < HW) qb[(bh * HW + n) * HD + d] = (__bf16)(v * 0.125f);  // scale folded
    } else if (sec == 1) {
      kb[(bh * NSEQ + n) * HD + d] = (__bf16)v;
    } else {
      vT[(bh * HD + d) * NSEQ + n] = (__bf16)v;                        // transposed
    }
  });
}

// ---- projection GEMM: [4608x768] x [768x768] + bias -> fp32 into d_out ----
__global__ void gemm_proj(const __bf16* __restrict__ A, const __bf16* __restrict__ W,
                          const float* __restrict__ bias, float* __restrict__ out) {
  const int row0 = blockIdx.x * 64, col0 = blockIdx.y * 128;
  gemm_64x128(A, W, row0, col0, [&](int rg, int cg, float v) {
    const int b_ = rg / HW, q = rg % HW;
    out[((size_t)b_ * NSEQ + q) * DIM + cg] = v + bias[cg];
  });
}

// ---- fused attention: scores -> per-frame top32 threshold -> softmax -> P@V ----
__global__ void attn_kernel(const __bf16* __restrict__ qb, const __bf16* __restrict__ kb,
                            const __bf16* __restrict__ vT, __bf16* __restrict__ ao) {
  extern __shared__ char smem[];
  float*  S = (float*)smem;                        // [16][2880] f32   (184320 B)
  __bf16* P = (__bf16*)(smem + 184320);            // [16][2880] bf16  ( 92160 B)
  __bf16* Q = (__bf16*)(smem + 276480);            // [16][64]   bf16  (  2048 B)
  const int qt = blockIdx.x, h = blockIdx.y, b = blockIdx.z;
  const int wave = threadIdx.x >> 5, lane = threadIdx.x & 31;
  const int m = lane & 15, half = lane >> 4;
  const size_t bh = (size_t)b * HEADS + h;

  // stage Q tile (16x64)
  if (threadIdx.x < 128) {
    const __bf16* src = qb + (bh * HW + qt * 16) * HD;
    reinterpret_cast<uint4*>(Q)[threadIdx.x] =
        reinterpret_cast<const uint4*>(src)[threadIdx.x];
  }
  __syncthreads();

  // ---- B1: S = (q*scale) @ K^T, WMMA over all 180 n-tiles ----
  const __bf16* qr = Q + m * HD;
  v16bf a0 = frag16(qr + half * 8,      qr + 16 + half * 8);
  v16bf a1 = frag16(qr + 32 + half * 8, qr + 48 + half * 8);
  const __bf16* kbase = kb + bh * (size_t)NSEQ * HD;
  for (int nt = wave; nt < NSEQ / 16; nt += 16) {
    const __bf16* kr = kbase + (size_t)(nt * 16 + m) * HD;
    if (nt + 16 < NSEQ / 16)
      __builtin_prefetch(kbase + (size_t)((nt + 16) * 16 + m) * HD, 0, 0);
    v16bf b0 = frag16(kr + half * 8,      kr + 16 + half * 8);
    v16bf b1 = frag16(kr + 32 + half * 8, kr + 48 + half * 8);
    v8f acc = {};
    acc = bwmma(a0, b0, acc);
    acc = bwmma(a1, b1, acc);
#pragma unroll
    for (int r = 0; r < 8; ++r)
      S[(r + 8 * half) * NSEQ + nt * 16 + m] = acc[r];
  }
  __syncthreads();

  // ---- B2: per-query (one wave each) top-32 threshold per frame + softmax ----
  {
    float* Sq = S + wave * NSEQ;
    float dm = -3.4e38f;
    for (int i = lane; i < HW; i += 32) dm = fmaxf(dm, Sq[i]);
    dm = wave_max(dm);
    float thr[TFR], fm[TFR];
#pragma unroll
    for (int f = 0; f < TFR; ++f) {
      const int base = HW + f * HW;
      float vals[18];
#pragma unroll
      for (int i = 0; i < 18; ++i) vals[i] = Sq[base + lane + 32 * i];
      unsigned alive = 0x3FFFFu;
      float wmax = -3.4e38f;
      for (int t = 0; t < 32; ++t) {
        float lm = -3.4e38f; int la = -1;
#pragma unroll
        for (int i = 0; i < 18; ++i)
          if (((alive >> i) & 1u) && vals[i] > lm) { lm = vals[i]; la = i; }
        wmax = wave_max(lm);
        unsigned long long bal = __ballot(lm == wmax);
        int owner = (int)__ffsll(bal) - 1;
        if (lane == owner) alive &= ~(1u << la);
        if (t == 0) fm[f] = wmax;
      }
      thr[f] = wmax;   // 32nd largest
    }
    float M = dm;
#pragma unroll
    for (int f = 0; f < TFR; ++f) M = fmaxf(M, fm[f]);
    float z = 0.f;
    for (int i = lane; i < HW; i += 32) z += __expf(Sq[i] - M);
#pragma unroll
    for (int f = 0; f < TFR; ++f) {
      const int base = HW + f * HW;
      for (int i = lane; i < HW; i += 32) {
        float v = Sq[base + i];
        if (v >= thr[f]) z += __expf(v - M);
      }
    }
    z = wave_sum(z);
    float inv = 1.f / z;
    __bf16* Pq = P + wave * NSEQ;
    for (int i = lane; i < HW; i += 32)
      Pq[i] = (__bf16)(__expf(Sq[i] - M) * inv);
#pragma unroll
    for (int f = 0; f < TFR; ++f) {
      const int base = HW + f * HW;
      for (int i = lane; i < HW; i += 32) {
        float v = Sq[base + i];
        Pq[base + i] = (v >= thr[f]) ? (__bf16)(__expf(v - M) * inv) : (__bf16)0.f;
      }
    }
  }
  __syncthreads();

  // ---- B3: out = P @ V ; 4 n-tiles x 4 k-groups over 16 waves ----
  const int ntile = wave & 3, kg = wave >> 2;
  const __bf16* vbase = vT + bh * (size_t)HD * NSEQ;
  v8f acc = {};
  for (int s = kg; s < NSEQ / 32; s += 4) {
    const int k0 = s * 32;
    const __bf16* pr = P + m * NSEQ + k0;
    v16bf a = frag16(pr + half * 8, pr + 16 + half * 8);
    const __bf16* vr = vbase + (size_t)(ntile * 16 + m) * NSEQ + k0;
    v16bf bb = frag16(vr + half * 8, vr + 16 + half * 8);
    acc = bwmma(a, bb, acc);
  }
  float* part = S;  // reuse score region: [16 waves][16][16]
#pragma unroll
  for (int r = 0; r < 8; ++r)
    part[((wave * 16) + r + 8 * half) * 16 + m] = acc[r];
  __syncthreads();
  for (int e = threadIdx.x; e < 1024; e += 512) {
    int nt = e >> 8, mm = (e >> 4) & 15, nn = e & 15;
    float sum = 0.f;
#pragma unroll
    for (int g = 0; g < 4; ++g) sum += part[((g * 4 + nt) * 16 + mm) * 16 + nn];
    ao[(size_t)(b * HW + qt * 16 + mm) * DIM + h * HD + nt * 16 + nn] = (__bf16)sum;
  }
}

extern "C" void kernel_launch(void* const* d_in, const int* in_sizes, int n_in,
                              void* d_out, int out_size, void* d_ws, size_t ws_size,
                              hipStream_t stream) {
  const float* x      = (const float*)d_in[0];
  const float* qkv_w  = (const float*)d_in[1];
  const float* proj_w = (const float*)d_in[2];
  const float* proj_b = (const float*)d_in[3];
  char* ws = (char*)d_ws;
  __bf16* xb = (__bf16*)(ws + 0);           // 23040*768*2 = 35389440
  __bf16* wq = (__bf16*)(ws + 35389440);    //  2304*768*2 =  3538944
  __bf16* wp = (__bf16*)(ws + 38928384);    //   768*768*2 =  1179648
  __bf16* qb = (__bf16*)(ws + 40108032);    //  8*12*576*64*2 = 7077888
  __bf16* kb = (__bf16*)(ws + 47185920);    //  8*12*2880*64*2 = 35389440
  __bf16* vT = (__bf16*)(ws + 82575360);    //  same, transposed
  __bf16* ao = (__bf16*)(ws + 117964800);   //  4608*768*2 = 7077888
  float* out = (float*)d_out;

  cvt_bf16<<<(MROWS * DIM / 4 + 255) / 256, 256, 0, stream>>>(x, xb, MROWS * DIM / 4);
  cvt_bf16<<<(3 * DIM * DIM / 4 + 255) / 256, 256, 0, stream>>>(qkv_w, wq, 3 * DIM * DIM / 4);
  cvt_bf16<<<(DIM * DIM / 4 + 255) / 256, 256, 0, stream>>>(proj_w, wp, DIM * DIM / 4);

  gemm_qkv<<<dim3(MROWS / 64, 3 * DIM / 128), 256, 0, stream>>>(xb, wq, qb, kb, vT);

  // 184320 (S) + 92160 (P) + 2048 (Q) = 278528 B of LDS (CDNA5: 320KB/WGP)
  attn_kernel<<<dim3(HW / 16, HEADS, NB), 512, 278528, stream>>>(qb, kb, vT, ao);

  gemm_proj<<<dim3(PROWS / 64, DIM / 128), 256, 0, stream>>>(ao, wp, proj_b, out);

  copy_tail<<<(NB * TFR * HW * DIM / 4 + 255) / 256, 256, 0, stream>>>(x, out);
}
